// GroupedQueryAttention_24739011625359
// MI455X (gfx1250) — matrix operand order repaired and checked by reading, not compiled, not run on hardware
//
#include <hip/hip_runtime.h>
#include <hip/hip_bf16.h>

#define D_MODEL 2048
#define NUM_HEADS 16
#define NUM_KV 4
#define HD 128
#define SEQ 2048
#define KV_DIM (NUM_KV * HD)   // 512

typedef __attribute__((ext_vector_type(16))) __bf16 v16bf;
typedef __attribute__((ext_vector_type(8)))  float v8f;
typedef __attribute__((ext_vector_type(4)))  unsigned int v4u;
typedef __attribute__((ext_vector_type(8)))  int v8i;
typedef __attribute__((ext_vector_type(4)))  int v4i;

union FragA { v16bf v; unsigned int u[8]; };
union FragC { v8f  v; float f[8]; };

__device__ __forceinline__ unsigned short f2bf(float x) {
  union { float f; unsigned int u; } a; a.f = x;
  unsigned int r = a.u + 0x7FFFu + ((a.u >> 16) & 1u);   // RNE
  return (unsigned short)(r >> 16);
}
__device__ __forceinline__ float bf2f(unsigned short h) {
  union { float f; unsigned int u; } a; a.u = ((unsigned int)h) << 16;
  return a.f;
}
__device__ __forceinline__ unsigned int pack2bf(float lo, float hi) {
  return (unsigned int)f2bf(lo) | ((unsigned int)f2bf(hi) << 16);
}
__device__ __forceinline__ v8f wmma_bf16(const FragA& a, const FragA& b, v8f c) {
  return __builtin_amdgcn_wmma_f32_16x16x32_bf16(
      false, a.v, false, b.v, (short)0, c, false, false);
}

// low 32 bits of flat LDS address == workgroup-relative LDS byte offset
__device__ __forceinline__ unsigned lds_offset(const void* p) {
  return (unsigned)(unsigned long long)p;
}

// ---- Tensor Data Mover: 2D tile (data_size = 4B units), optional LDS padding ----
// pad_int encoding: pads after 2^(pad_int+1) dwords; pad_amt encoding: (pad_amt+1) dwords.
__device__ __forceinline__ void tdm_load_2d(unsigned lds_off, const void* gptr,
                                            unsigned tile_w_dw, unsigned tile_h,
                                            unsigned stride_dw,
                                            unsigned pad_int, unsigned pad_amt) {
  unsigned long long ga = (unsigned long long)gptr;
  v4u g0 = { 1u,                                   // count=1 (valid user descriptor)
             lds_off,                              // lds_addr (bytes)
             (unsigned)ga,                         // global_addr[31:0]
             (unsigned)((ga >> 32) & 0x01FFFFFFu) | 0x80000000u };  // addr[56:32] | type=2
  v8i g1;
  g1[0] = (int)((2u << 16)                         // data_size = 4 bytes
              | (1u << 20)                         // pad_enable
              | (pad_int << 22) | (pad_amt << 25));
  g1[1] = (int)((tile_w_dw & 0xFFFFu) << 16);      // tensor_dim0[15:0] (== tile_w)
  g1[2] = (int)((tile_h & 0xFFFFu) << 16);         // tensor_dim0[31:16]=0, tensor_dim1[15:0]
  g1[3] = (int)((tile_w_dw & 0xFFFFu) << 16);      // tensor_dim1[31:16]=0, tile_dim0
  g1[4] = (int)(tile_h & 0xFFFFu);                 // tile_dim1, tile_dim2=0
  g1[5] = (int)stride_dw;                          // tensor_dim0_stride[31:0]
  g1[6] = 0;                                       // stride[47:32]=0, dim1_stride lo=0
  g1[7] = 0;
  v4i z4 = {0, 0, 0, 0};
#if defined(__clang_major__) && __clang_major__ >= 23
  v8i z8 = {0, 0, 0, 0, 0, 0, 0, 0};
  __builtin_amdgcn_tensor_load_to_lds(g0, g1, z4, z4, z8, 0);
#else
  __builtin_amdgcn_tensor_load_to_lds(g0, g1, z4, z4, 0);
#endif
}

// ---------------- f32 -> bf16 conversion (row-major, for A matrices) ----------------
__global__ void cvt_f32_to_bf16_kernel(const float* __restrict__ in,
                                       unsigned short* __restrict__ out, int n) {
  int i = blockIdx.x * blockDim.x + threadIdx.x;
  if (i < n) out[i] = f2bf(in[i]);
}

// ---------------- f32 -> bf16, k-paired panel swizzle (for B matrices) -------------
// in: f32 [K][N] row-major.  out: per 32-row panel p, dword [kp 0..15][n 0..N-1]
// containing {B[32p+2kp][n], B[32p+2kp+1][n]} -> WMMA B-frag reads become dwords.
__global__ void cvt_swizzle_b_kernel(const float* __restrict__ in,
                                     unsigned short* __restrict__ out, int K, int N) {
  int i = blockIdx.x * blockDim.x + threadIdx.x;
  if (i >= K * N) return;
  int k = i / N;
  int n = i - k * N;
  int p = k >> 5, kk = k & 31;
  int kp = kk >> 1, half = kk & 1;
  out[((size_t)(p * 16 + kp) * N + n) * 2 + half] = f2bf(in[(size_t)k * N + n]);
}

// ---------------- GEMM: C(MxN) = A(MxK,bf16,row) * B(KxN,bf16,k-paired) ------------
// 256 threads = 8 waves (4x2), WG tile 128x64, wave tile 32x32, K-step 32.
// TDM double-buffered LDS staging; wave 0 issues loads one step ahead.
template<int OUT_BF16>
__global__ __launch_bounds__(256)
void gemm_bf16_kernel(const unsigned short* __restrict__ Abf,
                      const unsigned short* __restrict__ Bsw,
                      void* __restrict__ Cp, int M, int N, int K) {
  __shared__ unsigned int Alds[2][128 * 17];   // rows x 16 dwords (+1 pad via TDM)
  __shared__ unsigned int Blds[2][16 * 66];    // kp rows x 64 dwords (+2 pad via TDM)
  const int t = threadIdx.x;
  const int lane = t & 31;
  const int wave = t >> 5;
  const int wm = wave >> 1, wn = wave & 1;
  const int hs = lane >> 4, l16 = lane & 15;
  const int m0 = blockIdx.y * 128;
  const int n0 = blockIdx.x * 64;
  const int Kdw = K >> 1;
  const int nk = K >> 5;

  FragC acc[2][2];
#pragma unroll
  for (int a = 0; a < 2; ++a)
#pragma unroll
    for (int b = 0; b < 2; ++b)
#pragma unroll
      for (int r = 0; r < 8; ++r) acc[a][b].f[r] = 0.0f;

  if (wave == 0) {
    tdm_load_2d(lds_offset(&Alds[0][0]), Abf + (size_t)m0 * K, 16, 128, Kdw, 3, 0);
    tdm_load_2d(lds_offset(&Blds[0][0]), Bsw + (size_t)n0 * 2, 64, 16, N, 5, 1);
  }

  for (int kt = 0; kt < nk; ++kt) {
    const int buf = kt & 1;
    if (wave == 0) __builtin_amdgcn_s_wait_tensorcnt((short)0);
    __syncthreads();
    if (wave == 0 && kt + 1 < nk) {
      tdm_load_2d(lds_offset(&Alds[buf ^ 1][0]),
                  Abf + (size_t)m0 * K + (kt + 1) * 32, 16, 128, Kdw, 3, 0);
      tdm_load_2d(lds_offset(&Blds[buf ^ 1][0]),
                  Bsw + ((size_t)(kt + 1) * 16 * N + n0) * 2, 64, 16, N, 5, 1);
    }

    const unsigned int* Al = Alds[buf];
    const unsigned int* Bl = Blds[buf];
    FragA af[2], bfr[2];
#pragma unroll
    for (int mt = 0; mt < 2; ++mt) {
      int row = wm * 32 + mt * 16 + l16;
#pragma unroll
      for (int j = 0; j < 8; ++j) {
        int cd = (j & 3) + hs * 4 + ((j >> 2) * 8);   // ISA A-frag K pattern
        af[mt].u[j] = Al[row * 17 + cd];
      }
    }
#pragma unroll
    for (int nt = 0; nt < 2; ++nt) {
      int col = wn * 32 + nt * 16 + l16;
      int kpb = hs * 4;
#pragma unroll
      for (int v = 0; v < 8; ++v)
        bfr[nt].u[v] = Bl[(kpb + (v & 3) + (v >> 2) * 8) * 66 + col];
    }
#pragma unroll
    for (int mt = 0; mt < 2; ++mt)
#pragma unroll
      for (int nt = 0; nt < 2; ++nt)
        acc[mt][nt].v = wmma_bf16(af[mt], bfr[nt], acc[mt][nt].v);
  }

#pragma unroll
  for (int mt = 0; mt < 2; ++mt)
#pragma unroll
    for (int nt = 0; nt < 2; ++nt)
#pragma unroll
      for (int r = 0; r < 8; ++r) {
        int gr = m0 + wm * 32 + mt * 16 + r + 8 * hs;
        int gc = n0 + wn * 32 + nt * 16 + l16;
        if (OUT_BF16)
          ((unsigned short*)Cp)[(size_t)gr * N + gc] = f2bf(acc[mt][nt].f[r]);
        else
          ((float*)Cp)[(size_t)gr * N + gc] = acc[mt][nt].f[r];
      }
}

// ---------------- RoPE: [S][nh*128] -> rotated [nh][S][128] ----------------
__global__ void rope_kernel(const unsigned short* __restrict__ in,
                            unsigned short* __restrict__ out, int nheads) {
  int i = blockIdx.x * blockDim.x + threadIdx.x;
  int total = nheads * SEQ * 64;
  if (i >= total) return;
  int h = i / (SEQ * 64);
  int rem = i - h * (SEQ * 64);
  int s = rem >> 6;
  int j = rem & 63;
  int rowbase = s * (nheads * HD) + h * HD;
  float a = bf2f(in[rowbase + j]);
  float b = bf2f(in[rowbase + j + 64]);
  float inv = __expf(-(float)j * 0.14391156608f);   // ln(10000)/64
  float th = (float)s * inv;
  float sn, cs;
  __sincosf(th, &sn, &cs);
  size_t ob = ((size_t)h * SEQ + s) * HD;
  out[ob + j]      = f2bf(a * cs - b * sn);
  out[ob + j + 64] = f2bf(b * cs + a * sn);
}

// ------- V rearrange+swizzle: [S][KV*128] -> [KV][panel s/32][kp][128] k-paired -----
__global__ void rearrange_v_kernel(const unsigned short* __restrict__ in,
                                   unsigned short* __restrict__ out) {
  int i = blockIdx.x * blockDim.x + threadIdx.x;
  if (i >= NUM_KV * SEQ * HD) return;
  int h = i / (SEQ * HD);
  int rem = i - h * (SEQ * HD);
  int s = rem / HD;
  int d = rem - s * HD;
  int p = s >> 5, kk = s & 31;
  int kp = kk >> 1, half = kk & 1;
  out[((size_t)((h * (SEQ / 32) + p) * 16 + kp) * HD + d) * 2 + half] =
      in[(size_t)s * KV_DIM + h * HD + d];
}

// ---------------- Flash attention (causal, GQA), WMMA bf16 + TDM ----------------
// grid: (SEQ/64, NUM_HEADS), block 128 (4 waves); each wave owns 16 q rows.
__global__ __launch_bounds__(128)
void attention_kernel(const unsigned int* __restrict__ Qdw,      // [H][S][64dw]
                      const unsigned short* __restrict__ Kbf,    // [KV][S][128] row-major
                      const unsigned short* __restrict__ Vsw,    // [KV][S/32][16][128] k-paired
                      unsigned short* __restrict__ ctx) {        // [S][2048]
  __shared__ unsigned int Klds[2][32 * 65];   // key rows x 64 dwords (+1 pad via TDM)
  __shared__ unsigned int Vlds[2][16 * 129];  // kp rows x 128 dwords (+1 pad via TDM)
  __shared__ float Plds[4 * 16 * 33];         // per-wave 16x32 probs (+1 pad)
  const int t = threadIdx.x;
  const int lane = t & 31;
  const int wave = t >> 5;
  const int hs = lane >> 4, l16 = lane & 15;
  const int h = blockIdx.y;
  const int kvh = h >> 2;                     // NUM_HEADS / NUM_KV = 4
  const int qb = blockIdx.x;
  const int q0 = qb * 64 + wave * 16;
  const float scale = 0.08838834764831845f;   // 1/sqrt(128)
  float* Pw = Plds + wave * (16 * 33);
  const unsigned short* Kbase = Kbf + (size_t)kvh * SEQ * HD;
  const unsigned short* Vbase = Vsw + (size_t)kvh * (SEQ / 32) * 16 * HD;

  // Q fragments in registers: 4 chunks of 16x32 bf16
  FragA qf[4];
  {
    const unsigned int* qrow = Qdw + ((size_t)h * SEQ + (q0 + l16)) * 64;
#pragma unroll
    for (int ch = 0; ch < 4; ++ch)
#pragma unroll
      for (int j = 0; j < 8; ++j) {
        int cd = hs * 8 + j;
        qf[ch].u[j] = qrow[ch * 16 + cd];
      }
  }

  FragC acc[8];
#pragma unroll
  for (int d = 0; d < 8; ++d)
#pragma unroll
    for (int r = 0; r < 8; ++r) acc[d].f[r] = 0.0f;
  float mrun[8], lrun[8];
#pragma unroll
  for (int r = 0; r < 8; ++r) { mrun[r] = -1e30f; lrun[r] = 0.0f; }

  const int nkb = qb * 2 + 2;                 // causal: keys up to end of q block
  if (wave == 0) {
    tdm_load_2d(lds_offset(&Klds[0][0]), Kbase, 64, 32, 64, 5, 0);
    tdm_load_2d(lds_offset(&Vlds[0][0]), Vbase, 128, 16, 128, 6, 0);
  }

  for (int kb = 0; kb < nkb; ++kb) {
    const int buf = kb & 1;
    if (wave == 0) __builtin_amdgcn_s_wait_tensorcnt((short)0);
    __syncthreads();
    if (wave == 0 && kb + 1 < nkb) {
      tdm_load_2d(lds_offset(&Klds[buf ^ 1][0]),
                  Kbase + (size_t)(kb + 1) * 32 * HD, 64, 32, 64, 5, 0);
      tdm_load_2d(lds_offset(&Vlds[buf ^ 1][0]),
                  Vbase + (size_t)(kb + 1) * 16 * HD * 2, 128, 16, 128, 6, 0);
    }
    const unsigned int* Kl = Klds[buf];
    const unsigned int* Vl = Vlds[buf];

    // scores: S = Q(16x128) @ K^T(128x32)  -> two 16x16 f32 tiles
    FragC s0, s1;
#pragma unroll
    for (int r = 0; r < 8; ++r) { s0.f[r] = 0.0f; s1.f[r] = 0.0f; }
#pragma unroll
    for (int ch = 0; ch < 4; ++ch) {
      FragA b0, b1;
#pragma unroll
      for (int v = 0; v < 8; ++v) {
        b0.u[v] = Kl[l16 * 65        + ch * 16 + hs * 8 + v];
        b1.u[v] = Kl[(16 + l16) * 65 + ch * 16 + hs * 8 + v];
      }
      s0.v = wmma_bf16(qf[ch], b0, s0.v);
      s1.v = wmma_bf16(qf[ch], b1, s1.v);
    }

    // online softmax (per q row) + write P to wave-private LDS
#pragma unroll
    for (int r = 0; r < 8; ++r) {
      int qi = q0 + r + 8 * hs;
      int k0i = kb * 32 + l16;
      float a0 = (k0i <= qi)      ? s0.f[r] * scale : -1e30f;
      float a1 = (k0i + 16 <= qi) ? s1.f[r] * scale : -1e30f;
      float rm = fmaxf(a0, a1);
#pragma unroll
      for (int m = 1; m <= 8; m <<= 1) rm = fmaxf(rm, __shfl_xor(rm, m, 32));
      float nm = fmaxf(mrun[r], rm);
      float p0 = __expf(a0 - nm);
      float p1 = __expf(a1 - nm);
      float rs = p0 + p1;
#pragma unroll
      for (int m = 1; m <= 8; m <<= 1) rs += __shfl_xor(rs, m, 32);
      float corr = __expf(mrun[r] - nm);
      lrun[r] = lrun[r] * corr + rs;
      mrun[r] = nm;
#pragma unroll
      for (int d = 0; d < 8; ++d) acc[d].f[r] *= corr;
      int mrow = r + 8 * hs;
      Pw[mrow * 33 + l16]      = p0;
      Pw[mrow * 33 + 16 + l16] = p1;
    }
    asm volatile("s_wait_dscnt 0x0" ::: "memory");   // wave-private LDS RAW

    // re-layout P into A-fragment (16x32) as bf16
    FragA pf;
#pragma unroll
    for (int j = 0; j < 8; ++j) {
      int kbase = ((j >> 2) * 16) + hs * 8 + (j & 3) * 2;
      pf.u[j] = pack2bf(Pw[l16 * 33 + kbase], Pw[l16 * 33 + kbase + 1]);
    }
    // PV: acc(16x128) += P(16x32) @ V(32x128); V frags are single dword reads
#pragma unroll
    for (int d = 0; d < 8; ++d) {
      FragA vf;
      int dcol = d * 16 + l16;
      int kpb = hs * 4;
#pragma unroll
      for (int v = 0; v < 8; ++v)
        vf.u[v] = Vl[(kpb + (v & 3) + (v >> 2) * 8) * 129 + dcol];
      acc[d].v = wmma_bf16(pf, vf, acc[d].v);
    }
  }

  // epilogue: normalize and write context [S][H*hd] bf16
#pragma unroll
  for (int d = 0; d < 8; ++d)
#pragma unroll
    for (int r = 0; r < 8; ++r) {
      int qi = q0 + r + 8 * hs;
      int col = h * HD + d * 16 + l16;
      ctx[(size_t)qi * D_MODEL + col] = f2bf(acc[d].f[r] / lrun[r]);
    }
}

extern "C" void kernel_launch(void* const* d_in, const int* in_sizes, int n_in,
                              void* d_out, int out_size, void* d_ws, size_t ws_size,
                              hipStream_t stream) {
  const float* H  = (const float*)d_in[0];
  // d_in[1] = attention_mask: causal -> computed analytically in-kernel
  const float* Wq = (const float*)d_in[2];
  const float* Wk = (const float*)d_in[3];
  const float* Wv = (const float*)d_in[4];
  const float* Wo = (const float*)d_in[5];

  char* ws = (char*)d_ws;
  size_t off = 0;
  auto alloc = [&](size_t bytes) {
    char* p = ws + off;
    off += (bytes + 255) & ~(size_t)255;
    return p;
  };
  unsigned short* Hbf = (unsigned short*)alloc((size_t)SEQ * D_MODEL * 2);
  unsigned short* Wqb = (unsigned short*)alloc((size_t)D_MODEL * D_MODEL * 2);
  unsigned short* Wkb = (unsigned short*)alloc((size_t)D_MODEL * KV_DIM * 2);
  unsigned short* Wvb = (unsigned short*)alloc((size_t)D_MODEL * KV_DIM * 2);
  unsigned short* Wob = (unsigned short*)alloc((size_t)D_MODEL * D_MODEL * 2);
  unsigned short* Qg  = (unsigned short*)alloc((size_t)SEQ * D_MODEL * 2);
  unsigned short* Kg  = (unsigned short*)alloc((size_t)SEQ * KV_DIM * 2);
  unsigned short* Vg  = (unsigned short*)alloc((size_t)SEQ * KV_DIM * 2);
  unsigned short* Qh  = (unsigned short*)alloc((size_t)SEQ * D_MODEL * 2);
  unsigned short* Kh  = (unsigned short*)alloc((size_t)SEQ * KV_DIM * 2);
  unsigned short* Vh  = (unsigned short*)alloc((size_t)SEQ * KV_DIM * 2);
  unsigned short* Ctx = (unsigned short*)alloc((size_t)SEQ * D_MODEL * 2);

  // A-side (row-major) conversion
  cvt_f32_to_bf16_kernel<<<(SEQ * D_MODEL + 255) / 256, 256, 0, stream>>>(
      H, Hbf, SEQ * D_MODEL);
  // B-side conversions with k-paired panel swizzle
  cvt_swizzle_b_kernel<<<(D_MODEL * D_MODEL + 255) / 256, 256, 0, stream>>>(
      Wq, Wqb, D_MODEL, D_MODEL);
  cvt_swizzle_b_kernel<<<(D_MODEL * KV_DIM + 255) / 256, 256, 0, stream>>>(
      Wk, Wkb, D_MODEL, KV_DIM);
  cvt_swizzle_b_kernel<<<(D_MODEL * KV_DIM + 255) / 256, 256, 0, stream>>>(
      Wv, Wvb, D_MODEL, KV_DIM);
  cvt_swizzle_b_kernel<<<(D_MODEL * D_MODEL + 255) / 256, 256, 0, stream>>>(
      Wo, Wob, D_MODEL, D_MODEL);

  // QKV projections (bf16 out)
  gemm_bf16_kernel<1><<<dim3(D_MODEL / 64, SEQ / 128), 256, 0, stream>>>(
      Hbf, Wqb, Qg, SEQ, D_MODEL, D_MODEL);
  gemm_bf16_kernel<1><<<dim3(KV_DIM / 64, SEQ / 128), 256, 0, stream>>>(
      Hbf, Wkb, Kg, SEQ, KV_DIM, D_MODEL);
  gemm_bf16_kernel<1><<<dim3(KV_DIM / 64, SEQ / 128), 256, 0, stream>>>(
      Hbf, Wvb, Vg, SEQ, KV_DIM, D_MODEL);

  // RoPE + head-major rearrange (+ V k-paired swizzle)
  rope_kernel<<<(NUM_HEADS * SEQ * 64 + 255) / 256, 256, 0, stream>>>(Qg, Qh, NUM_HEADS);
  rope_kernel<<<(NUM_KV * SEQ * 64 + 255) / 256, 256, 0, stream>>>(Kg, Kh, NUM_KV);
  rearrange_v_kernel<<<(NUM_KV * SEQ * HD + 255) / 256, 256, 0, stream>>>(Vg, Vh);

  // causal GQA flash attention
  attention_kernel<<<dim3(SEQ / 64, NUM_HEADS), 128, 0, stream>>>(
      (const unsigned int*)Qh, Kh, Vh, Ctx);

  // output projection (f32 out, straight to d_out)
  gemm_bf16_kernel<0><<<dim3(D_MODEL / 64, SEQ / 128), 256, 0, stream>>>(
      Ctx, Wob, d_out, SEQ, D_MODEL, D_MODEL);
}
